// MoE_57389353009460
// MI455X (gfx1250) — matrix operand order repaired and checked by reading, not compiled, not run on hardware
//
#include <hip/hip_runtime.h>
#include <hip/hip_bf16.h>
#include <math.h>

typedef __bf16 bf16;
typedef __attribute__((ext_vector_type(16))) __bf16 v16bf;
typedef __attribute__((ext_vector_type(8)))  __bf16 v8bf;
typedef __attribute__((ext_vector_type(4)))  __bf16 v4bf;
typedef __attribute__((ext_vector_type(8)))  float  v8f;
typedef __attribute__((ext_vector_type(4)))  unsigned int v4u;
typedef __attribute__((ext_vector_type(8)))  int v8i;
typedef __attribute__((ext_vector_type(4)))  int v4i;

#define T_TOK 4096
#define DDIM  1024
#define HDIM  1024
#define NEXP  8
#define TOPK  2
#define TCAP  4096   // worst-case tokens per expert

// GEMM tiling: 256 threads = 8 waves (2x4), block tile 64M x 128N, K-step 32
#define BM 64
#define BN 128
#define BK 32
#define LDA 40   // BK + 8 halves pad (80B rows, 16B aligned)
#define LDB 40   // B stored [n][k], BK + 8 halves pad

// ---------------------------------------------------------------- gating ----
__global__ void moe_gating(const float* __restrict__ x, const float* __restrict__ wg,
                           int* __restrict__ topi, float* __restrict__ topv) {
  const int wave = threadIdx.x >> 5;
  const int lane = threadIdx.x & 31;
  const int t = blockIdx.x * 8 + wave;
  if (t >= T_TOK) return;

  float acc[NEXP];
#pragma unroll
  for (int e = 0; e < NEXP; ++e) acc[e] = 0.0f;

  const float* xr = x + (size_t)t * DDIM;
  for (int d = lane; d < DDIM; d += 32) {
    float xv = xr[d];
    const float4* wr = (const float4*)(wg + (size_t)d * NEXP);
    float4 w0 = wr[0], w1 = wr[1];
    acc[0] += xv * w0.x; acc[1] += xv * w0.y; acc[2] += xv * w0.z; acc[3] += xv * w0.w;
    acc[4] += xv * w1.x; acc[5] += xv * w1.y; acc[6] += xv * w1.z; acc[7] += xv * w1.w;
  }
#pragma unroll
  for (int e = 0; e < NEXP; ++e) {
#pragma unroll
    for (int off = 16; off >= 1; off >>= 1)
      acc[e] += __shfl_xor(acc[e], off, 32);
  }
  if (lane == 0) {
    float m = acc[0];
#pragma unroll
    for (int e = 1; e < NEXP; ++e) m = fmaxf(m, acc[e]);
    float p[NEXP], s = 0.0f;
#pragma unroll
    for (int e = 0; e < NEXP; ++e) { p[e] = __expf(acc[e] - m); s += p[e]; }
    float inv = 1.0f / s;
#pragma unroll
    for (int e = 0; e < NEXP; ++e) p[e] *= inv;
    int i0 = 0; float v0 = p[0];
#pragma unroll
    for (int e = 1; e < NEXP; ++e) if (p[e] > v0) { v0 = p[e]; i0 = e; }
    int i1 = -1; float v1 = -1.0f;
#pragma unroll
    for (int e = 0; e < NEXP; ++e) if (e != i0 && p[e] > v1) { v1 = p[e]; i1 = e; }
    float den = v0 + v1 + 1e-6f;
    topi[t * 2 + 0] = i0; topi[t * 2 + 1] = i1;
    topv[t * 2 + 0] = v0 / den; topv[t * 2 + 1] = v1 / den;
  }
}

// ------------------------------------------- deterministic expert lists ----
__global__ void moe_build(const int* __restrict__ topi, const float* __restrict__ topv,
                          int* __restrict__ rowTok, float* __restrict__ rowGate,
                          int* __restrict__ counts, int* __restrict__ offsets) {
  __shared__ int totals[NEXP];
  const int e = threadIdx.x >> 5;       // one wave per expert
  const int lane = threadIdx.x & 31;
  int total = 0;
  const int NA = T_TOK * TOPK;          // 8192, multiple of 32
  for (int base = 0; base < NA; base += 32) {
    int a = base + lane;
    int ea = topi[a];
    bool match = (ea == e);
    unsigned mask = (unsigned)__ballot(match ? 1 : 0);
    int rank = __popc(mask & ((1u << lane) - 1u));
    if (match) {
      rowTok[e * TCAP + total + rank]  = a >> 1;   // token id
      rowGate[e * TCAP + total + rank] = topv[a];
    }
    total += __popc(mask);
  }
  if (lane == 0) { counts[e] = total; totals[e] = total; }
  __syncthreads();
  if (threadIdx.x == 0) {
    int off = 0;
    for (int i = 0; i < NEXP; ++i) { offsets[i] = off; off += totals[i]; }
  }
}

// ---------------------------------------------------------------- zero y ----
__global__ void moe_zero(float4* __restrict__ y, int n4) {
  int i = blockIdx.x * blockDim.x + threadIdx.x;
  if (i < n4) y[i] = make_float4(0.f, 0.f, 0.f, 0.f);
}

// --------------------------------------------------------- WMMA helpers ----
__device__ __forceinline__ v16bf load_fragA(const bf16* lds, int row, int hi) {
  const bf16* pa = lds + row * LDA + hi * 8;        // K = hi*8.. and 16+hi*8..
  v8bf lo = *(const v8bf*)pa;
  v8bf hi8 = *(const v8bf*)(pa + 16);
  v16bf a;
#pragma unroll
  for (int i = 0; i < 8; ++i) { a[i] = lo[i]; a[8 + i] = hi8[i]; }
  return a;
}
__device__ __forceinline__ v16bf load_fragB(const bf16* lds, int n, int hi) {
  const bf16* pb = lds + n * LDB + hi * 8;          // K = hi*8.. and 16+hi*8..
  v8bf b0 = *(const v8bf*)pb;
  v8bf b1 = *(const v8bf*)(pb + 16);
  v16bf b;
#pragma unroll
  for (int i = 0; i < 8; ++i) { b[i] = b0[i]; b[8 + i] = b1[i]; }
  return b;
}

// ------------------------------------------------ staging (reg pipeline) ----
struct Bregs { float4 v[4]; };

__device__ __forceinline__ Bregs loadB(const float* src, int ldn) {
  Bregs r;
#pragma unroll
  for (int i = 0; i < 4; ++i) r.v[i] = *(const float4*)(src + (size_t)i * ldn);
  return r;
}
// thread owns k = bkq*4..+3, n = bng*4..+3 ; store transposed [n][k] as 4x b64
__device__ __forceinline__ void storeB(bf16* ldsB, const Bregs& r, int bkq, int bng) {
  bf16* base = &ldsB[(bng * 4) * LDB + bkq * 4];
  v4bf t0 = (v4bf){(bf16)r.v[0].x, (bf16)r.v[1].x, (bf16)r.v[2].x, (bf16)r.v[3].x};
  v4bf t1 = (v4bf){(bf16)r.v[0].y, (bf16)r.v[1].y, (bf16)r.v[2].y, (bf16)r.v[3].y};
  v4bf t2 = (v4bf){(bf16)r.v[0].z, (bf16)r.v[1].z, (bf16)r.v[2].z, (bf16)r.v[3].z};
  v4bf t3 = (v4bf){(bf16)r.v[0].w, (bf16)r.v[1].w, (bf16)r.v[2].w, (bf16)r.v[3].w};
  *(v4bf*)(base)           = t0;
  *(v4bf*)(base + LDB)     = t1;
  *(v4bf*)(base + 2 * LDB) = t2;
  *(v4bf*)(base + 3 * LDB) = t3;
}
__device__ __forceinline__ void storeA8(bf16* dst, float4 f0, float4 f1) {
  v8bf t = (v8bf){(bf16)f0.x, (bf16)f0.y, (bf16)f0.z, (bf16)f0.w,
                  (bf16)f1.x, (bf16)f1.y, (bf16)f1.z, (bf16)f1.w};
  *(v8bf*)dst = t;  // single ds_store_b128
}

// --------------------------- TDM: async 2D bf16 tile -> padded LDS rows ----
// tile: BM rows x BK cols of 2-byte elements, row stride = ldn elements,
// LDS rows padded to LDA halves (pad_interval 16 dwords, pad_amount 4 dwords)
__device__ __forceinline__ void tdm_load_tileA(unsigned lds_off, const bf16* gptr,
                                               int rows_avail, int ldn) {
  unsigned long long ga = (unsigned long long)(uintptr_t)gptr;
  v4u g0;
  g0[0] = 1u;                                            // count=1, user desc
  g0[1] = lds_off;                                       // lds_addr (bytes)
  g0[2] = (unsigned)(ga & 0xffffffffu);                  // global_addr[31:0]
  g0[3] = (unsigned)((ga >> 32) & 0x01ffffffu) | 0x80000000u; // [56:32] | type=2
  unsigned dim0 = (unsigned)ldn;                         // tensor_dim0 (elements)
  unsigned dim1 = (unsigned)(rows_avail < 0 ? 0 : rows_avail); // clamp -> OOB rows read 0
  v8i g1;
  g1[0] = (int)((1u << 16)        // data_size = 2 bytes
              | (1u << 20)        // pad_enable
              | (3u << 22)        // pad_interval: 16 dwords (one 64B tile row)
              | (3u << 25));      // pad_amount: 4 dwords (8 halves)
  g1[1] = (int)((dim0 & 0xffffu) << 16);                        // dim0[15:0]
  g1[2] = (int)((dim0 >> 16) | ((dim1 & 0xffffu) << 16));       // dim0[31:16]|dim1[15:0]
  g1[3] = (int)((dim1 >> 16) | ((unsigned)BK << 16));           // dim1[31:16]|tile_dim0
  g1[4] = (int)BM;                                              // tile_dim1 (tile_dim2=0)
  g1[5] = (int)(unsigned)ldn;                                   // tensor_dim0_stride[31:0]
  g1[6] = 0;
  g1[7] = 0;
  v4i gz4 = (v4i){0, 0, 0, 0};
  v8i gz8 = (v8i){0, 0, 0, 0, 0, 0, 0, 0};
  // 6-arg form (clang-23 / therock headers): g0, g1, g2, g3, g4, cpol
  __builtin_amdgcn_tensor_load_to_lds(g0, g1, gz4, gz4, gz8, 0);
}

// ------------------------------------------------- GEMM1: x@w1+b1 -> GELU ----
__global__ void __launch_bounds__(256) moe_gemm1(
    const float* __restrict__ x, const float* __restrict__ w1, const float* __restrict__ b1,
    const int* __restrict__ rowTok, const int* __restrict__ counts,
    const int* __restrict__ offsets, bf16* __restrict__ act) {
  const int e = blockIdx.z;
  const int cnt = counts[e];
  const int rowbase = blockIdx.y * BM;
  if (rowbase >= cnt) return;
  const int n0 = blockIdx.x * BN;
  const int aoff = offsets[e];

  __shared__ __align__(16) bf16 ldsA[BM * LDA];
  __shared__ __align__(16) bf16 ldsB[BN * LDB];
  __shared__ int tokLds[BM];

  const int tid = threadIdx.x;
  if (tid < BM) {
    int r = rowbase + tid;
    int rc = (r < cnt) ? r : (cnt - 1);                 // clamp: dup a valid row
    tokLds[tid] = rowTok[e * TCAP + rc];
  }
  __syncthreads();

  const int wave = tid >> 5, lane = tid & 31;
  const int wm = wave >> 2, wn = wave & 3;              // 2x4 wave grid
  const int lr = lane & 15, hi = lane >> 4;

  const int ar = tid >> 2, acq = tid & 3;               // A: row, col-group of 8
  const int bkq = tid >> 5, bng = tid & 31;             // B: 4k x 4n sub-tile
  const int tokr = tokLds[ar];
  const float* aptr = x + (size_t)tokr * DDIM + acq * 8;
  const float* bptr = w1 + (size_t)e * DDIM * HDIM + (size_t)(bkq * 4) * HDIM + n0 + bng * 4;

  float4 a0 = *(const float4*)(aptr);
  float4 a1 = *(const float4*)(aptr + 4);
  Bregs br = loadB(bptr, HDIM);

  v8f c[2][2];
#pragma unroll
  for (int i = 0; i < 2; ++i)
#pragma unroll
    for (int j = 0; j < 2; ++j) c[i][j] = (v8f){0.f,0.f,0.f,0.f,0.f,0.f,0.f,0.f};

  for (int k0 = 0; k0 < DDIM; k0 += BK) {
    storeA8(&ldsA[ar * LDA + acq * 8], a0, a1);
    storeB(ldsB, br, bkq, bng);
    __syncthreads();

    const bool more = (k0 + BK) < DDIM;
    if (more) {  // prefetch next tiles into registers while WMMAs run
      a0 = *(const float4*)(aptr + k0 + BK);
      a1 = *(const float4*)(aptr + k0 + BK + 4);
      br = loadB(bptr + (size_t)(k0 + BK) * HDIM, HDIM);
    }

    v16bf a[2], b[2];
#pragma unroll
    for (int ti = 0; ti < 2; ++ti) a[ti] = load_fragA(ldsA, wm * 32 + ti * 16 + lr, hi);
#pragma unroll
    for (int tj = 0; tj < 2; ++tj) b[tj] = load_fragB(ldsB, wn * 32 + tj * 16 + lr, hi);
#pragma unroll
    for (int ti = 0; ti < 2; ++ti)
#pragma unroll
      for (int tj = 0; tj < 2; ++tj)
        c[ti][tj] = __builtin_amdgcn_wmma_f32_16x16x32_bf16(
            false, a[ti], false, b[tj], (short)0, c[ti][tj], false, false);
    __syncthreads();
  }

  // epilogue: + bias, exact-erf GELU, store bf16 activations
#pragma unroll
  for (int ti = 0; ti < 2; ++ti) {
#pragma unroll
    for (int tj = 0; tj < 2; ++tj) {
      int col = n0 + wn * 32 + tj * 16 + lr;
      float bias = b1[e * HDIM + col];
#pragma unroll
      for (int i = 0; i < 8; ++i) {
        int rg = rowbase + wm * 32 + ti * 16 + hi * 8 + i;
        if (rg < cnt) {
          float h = c[ti][tj][i] + bias;
          float g = 0.5f * h * (1.0f + erff(h * 0.70710678118654752f));
          act[(size_t)(aoff + rg) * HDIM + col] = (bf16)g;
        }
      }
    }
  }
}

// ------------------------------- GEMM2: act@w2+b2, *gate, scatter-add y ----
__global__ void __launch_bounds__(256) moe_gemm2(
    const bf16* __restrict__ act, const float* __restrict__ w2, const float* __restrict__ b2,
    const int* __restrict__ rowTok, const float* __restrict__ rowGate,
    const int* __restrict__ counts, const int* __restrict__ offsets,
    float* __restrict__ y) {
  const int e = blockIdx.z;
  const int cnt = counts[e];
  const int rowbase = blockIdx.y * BM;
  if (rowbase >= cnt) return;
  const int n0 = blockIdx.x * BN;
  const int aoff = offsets[e];

  __shared__ __align__(16) bf16 ldsA[BM * LDA];
  __shared__ __align__(16) bf16 ldsB[BN * LDB];
  __shared__ int tokLds[BM];
  __shared__ float gateLds[BM];

  const int tid = threadIdx.x;
  if (tid < BM) {
    int r = rowbase + tid;
    tokLds[tid]  = (r < cnt) ? rowTok[e * TCAP + r] : 0;
    gateLds[tid] = (r < cnt) ? rowGate[e * TCAP + r] : 0.0f;
  }

  const int wave = tid >> 5, lane = tid & 31;
  const int wm = wave >> 2, wn = wave & 3;
  const int lr = lane & 15, hi = lane >> 4;
  const int bkq = tid >> 5, bng = tid & 31;

  const bf16* abase = act + (size_t)(aoff + rowbase) * HDIM;   // contiguous rows
  const int rows_avail = cnt - rowbase;
  const unsigned ldsA_off = (unsigned)(uintptr_t)(&ldsA[0]);   // LDS byte offset

  // async DMA of first A tile (wave 0 issues; EXEC-independent, wave-level op)
  if (tid < 32) tdm_load_tileA(ldsA_off, abase, rows_avail, HDIM);

  const float* bptr = w2 + (size_t)e * HDIM * DDIM + (size_t)(bkq * 4) * DDIM + n0 + bng * 4;
  Bregs br = loadB(bptr, DDIM);

  v8f c[2][2];
#pragma unroll
  for (int i = 0; i < 2; ++i)
#pragma unroll
    for (int j = 0; j < 2; ++j) c[i][j] = (v8f){0.f,0.f,0.f,0.f,0.f,0.f,0.f,0.f};

  for (int k0 = 0; k0 < HDIM; k0 += BK) {
    storeB(ldsB, br, bkq, bng);
    if (tid < 32) __builtin_amdgcn_s_wait_tensorcnt((short)0);  // A tile landed
    __syncthreads();

    const bool more = (k0 + BK) < HDIM;
    if (more) br = loadB(bptr + (size_t)(k0 + BK) * DDIM, DDIM);

    v16bf a[2], b[2];
#pragma unroll
    for (int ti = 0; ti < 2; ++ti) a[ti] = load_fragA(ldsA, wm * 32 + ti * 16 + lr, hi);
#pragma unroll
    for (int tj = 0; tj < 2; ++tj) b[tj] = load_fragB(ldsB, wn * 32 + tj * 16 + lr, hi);
#pragma unroll
    for (int ti = 0; ti < 2; ++ti)
#pragma unroll
      for (int tj = 0; tj < 2; ++tj)
        c[ti][tj] = __builtin_amdgcn_wmma_f32_16x16x32_bf16(
            false, a[ti], false, b[tj], (short)0, c[ti][tj], false, false);
    __syncthreads();

    if (more && tid < 32)
      tdm_load_tileA(ldsA_off, abase + (k0 + BK), rows_avail, HDIM);
  }

  // epilogue: (c + b2) * gate, atomic scatter into y
#pragma unroll
  for (int ti = 0; ti < 2; ++ti) {
#pragma unroll
    for (int tj = 0; tj < 2; ++tj) {
      int col = n0 + wn * 32 + tj * 16 + lr;
      float bias = b2[e * DDIM + col];
#pragma unroll
      for (int i = 0; i < 8; ++i) {
        int r = wm * 32 + ti * 16 + hi * 8 + i;
        int rg = rowbase + r;
        if (rg < cnt) {
          float v = (c[ti][tj][i] + bias) * gateLds[r];
          unsafeAtomicAdd(&y[(size_t)tokLds[r] * DDIM + col], v);  // global_atomic_add_f32
        }
      }
    }
  }
}

// ----------------------------------------------------------------- launch ----
extern "C" void kernel_launch(void* const* d_in, const int* in_sizes, int n_in,
                              void* d_out, int out_size, void* d_ws, size_t ws_size,
                              hipStream_t stream) {
  const float* x  = (const float*)d_in[0];
  const float* w1 = (const float*)d_in[1];
  const float* b1 = (const float*)d_in[2];
  const float* w2 = (const float*)d_in[3];
  const float* b2 = (const float*)d_in[4];
  const float* wg = (const float*)d_in[5];
  float* y = (float*)d_out;

  // workspace partition (16B aligned slices)
  char* p = (char*)d_ws;
  int*   topi    = (int*)p;    p += (size_t)T_TOK * TOPK * sizeof(int);
  float* topv    = (float*)p;  p += (size_t)T_TOK * TOPK * sizeof(float);
  int*   rowTok  = (int*)p;    p += (size_t)NEXP * TCAP * sizeof(int);
  float* rowGate = (float*)p;  p += (size_t)NEXP * TCAP * sizeof(float);
  int*   counts  = (int*)p;    p += 256;
  int*   offsets = (int*)p;    p += 256;
  bf16*  act     = (bf16*)p;   // T_TOK*TOPK*HDIM bf16 = 16 MB

  moe_gating<<<dim3(T_TOK / 8), 256, 0, stream>>>(x, wg, topi, topv);
  moe_build<<<dim3(1), 256, 0, stream>>>(topi, topv, rowTok, rowGate, counts, offsets);
  moe_zero<<<dim3((T_TOK * DDIM / 4 + 255) / 256), 256, 0, stream>>>((float4*)y, T_TOK * DDIM / 4);
  moe_gemm1<<<dim3(HDIM / BN, TCAP / BM, NEXP), 256, 0, stream>>>(
      x, w1, b1, rowTok, counts, offsets, act);
  moe_gemm2<<<dim3(DDIM / BN, TCAP / BM, NEXP), 256, 0, stream>>>(
      act, w2, b2, rowTok, rowGate, counts, offsets, y);
}
